// WSConv2d_37151467110913
// MI455X (gfx1250) — matrix operand order repaired and checked by reading, not compiled, not run on hardware
//
#include <hip/hip_runtime.h>

// ---------------------------------------------------------------------------
// Types for WMMA fragments (CDNA5 / gfx1250, wave32)
// ---------------------------------------------------------------------------
typedef __attribute__((ext_vector_type(16))) __bf16        v16bf;
typedef __attribute__((ext_vector_type(8)))  float         v8f;
typedef __attribute__((ext_vector_type(4)))  int           i32x4;
typedef __attribute__((ext_vector_type(8)))  int           i32x8;
typedef __attribute__((ext_vector_type(4)))  unsigned int  u32x4;
typedef __attribute__((ext_vector_type(4)))  float         f32x4;

union Frag {
  i32x4 q[2];   // two 16-byte loads
  v16bf v;      // 16 bf16 per lane = one A/B fragment operand
};

#define BATCH   32
#define C_TOT   256                  // C_IN*G_IN == C_OUT*G_OUT
#define HW      64
#define IMG     (C_TOT * HW * HW)    // 1048576 elems per batch image
#define KTAPS   9
#define XSTRIDE 264                  // 256 channels + 8 pad (LDS bank spread)
#define PADH    66                   // 64 + halo
#define SLAB_ELEMS (PADH * XSTRIDE)  // one padded row: 17424 bf16 = 34848 B
#define SLAB_BYTES (SLAB_ELEMS * 2)
#define LDS_ELEMS (3 * SLAB_ELEMS)   // 3 padded rows
#define LDS_BYTES (LDS_ELEMS * 2)    // 104544 B  (WGP has 320 KB)
#define WB_OFF    32768
#define XPAD_OFF  (1u << 21)                               // 2 MB
#define XPAD_BYTES ((size_t)BATCH * PADH * SLAB_BYTES)     // 73.6 MB

__device__ __forceinline__ unsigned short f2bf(float f) {
  unsigned u = __float_as_uint(f);
  u += 0x7fffu + ((u >> 16) & 1u);   // round-to-nearest-even
  return (unsigned short)(u >> 16);
}

// ---------------------------------------------------------------------------
// Stage 1: Sinkhorn normalization of representations [4][36][36]
// ---------------------------------------------------------------------------
__global__ __launch_bounds__(64) void sinkhorn_kernel(const float* __restrict__ rep,
                                                      float* __restrict__ repOut) {
  const int g = blockIdx.x;
  const int t = threadIdx.x;
  __shared__ float la[36 * 36];
  __shared__ float lse[36];
  __shared__ int   flags;
  __shared__ int   doneFlag;

  for (int i = t; i < 1296; i += 64) la[i] = rep[g * 1296 + i];
  if (t == 0) doneFlag = 0;
  __syncthreads();

  for (int it = 0; it < 20; ++it) {
    if (t == 0) flags = 1;
    __syncthreads();
    if (t < 36) {
      float rs = 0.f, cs = 0.f;
      for (int m = 0; m < 36; ++m) {
        rs += expf(la[t * 36 + m]);
        cs += expf(la[m * 36 + t]);
      }
      if (fabsf(rs - 1.f) > 1.00001e-5f || fabsf(cs - 1.f) > 1.00001e-5f) flags = 0;
    }
    __syncthreads();
    if (t == 0 && flags) doneFlag = 1;
    __syncthreads();
    if (doneFlag) continue;          // uniform across block

    if (t < 36) {
      float mx = -INFINITY;
      for (int m = 0; m < 36; ++m) mx = fmaxf(mx, la[t * 36 + m]);
      float s = 0.f;
      for (int m = 0; m < 36; ++m) s += expf(la[t * 36 + m] - mx);
      lse[t] = mx + logf(s);
    }
    __syncthreads();
    if (t < 36) {
      float r = lse[t];
      for (int m = 0; m < 36; ++m) la[t * 36 + m] -= r;
    }
    __syncthreads();
    if (t < 36) {
      float mx = -INFINITY;
      for (int m = 0; m < 36; ++m) mx = fmaxf(mx, la[m * 36 + t]);
      float s = 0.f;
      for (int m = 0; m < 36; ++m) s += expf(la[m * 36 + t] - mx);
      lse[t] = mx + logf(s);
    }
    __syncthreads();
    if (t < 36) {
      for (int m = 0; m < 36; ++m) la[t * 36 + m] -= lse[m];
    }
    __syncthreads();
  }

  for (int i = t; i < 1296; i += 64) repOut[g * 1296 + i] = expf(la[i]);
}

// ---------------------------------------------------------------------------
// Stage 2: bake kernel into bf16 Wb[tap][oc][c] (WMMA-A-fragment-native)
// ---------------------------------------------------------------------------
__global__ __launch_bounds__(256) void build_weights_kernel(
    const float* __restrict__ rep,        // sinkhorned probs [4][36][36]
    const float* __restrict__ weight,     // [64][64][36]
    unsigned short* __restrict__ Wb) {    // [9][256][256] bf16
  const int oc = blockIdx.x;
  const int c  = threadIdx.x;
  const int o  = oc >> 2, g  = oc & 3;
  const int ci = c  >> 2, gi = c  & 3;
  const float* wrow = weight + (o * 64 + ci) * 36;
  const float* repg = rep + g * 1296;
#pragma unroll
  for (int tap = 0; tap < KTAPS; ++tap) {
    const int n = gi * 9 + tap;
    float acc = 0.f;
#pragma unroll
    for (int m = 0; m < 36; ++m) acc += repg[n * 36 + m] * wrow[m];
    Wb[(tap * 256 + oc) * 256 + c] = f2bf(acc);
  }
}

// ---------------------------------------------------------------------------
// Stage 2.5: build padded, pre-transposed bf16 activations
//   Xpad[b][row 0..65][p 0..65][c 0..263], row=y+1, p=x+1; halo/ch-pad = 0.
// One block per (b, padded row). Transpose is done ONCE here instead of ~6x
// per conv block. Output writes are fully contiguous per slab.
// ---------------------------------------------------------------------------
__global__ __launch_bounds__(256) void pad_x_bf16_kernel(
    const float* __restrict__ x, unsigned short* __restrict__ Xpad) {
  __shared__ unsigned short tile[SLAB_ELEMS];   // 34848 B
  const int bid = blockIdx.x;                   // BATCH * PADH
  const int row = bid % PADH;
  const int b   = bid / PADH;
  unsigned short* dst = Xpad + (size_t)bid * SLAB_ELEMS;
  const i32x4 z4 = {0, 0, 0, 0};

  if (row == 0 || row == PADH - 1) {            // zero halo rows
    for (int i = threadIdx.x; i < SLAB_BYTES / 16; i += 256)
      ((i32x4*)dst)[i] = z4;
    return;
  }
  const int yy = row - 1;

  for (int i = threadIdx.x; i < SLAB_BYTES / 16; i += 256)
    ((i32x4*)tile)[i] = z4;                     // covers p=0/65 and c>=256 pad
  __syncthreads();

  // coalesced reads: 16-lane groups share a channel, stream 64 floats
  for (int task = threadIdx.x; task < 256 * 16; task += 256) {
    const int c  = task >> 4;
    const int xc = (task & 15) * 4;
    f32x4 v = *(const f32x4*)(x + (size_t)b * IMG + (size_t)c * (HW * HW) +
                              (size_t)yy * HW + xc);
    tile[(xc + 1) * XSTRIDE + c] = f2bf(v.x);
    tile[(xc + 2) * XSTRIDE + c] = f2bf(v.y);
    tile[(xc + 3) * XSTRIDE + c] = f2bf(v.z);
    tile[(xc + 4) * XSTRIDE + c] = f2bf(v.w);
  }
  __syncthreads();

  for (int i = threadIdx.x; i < SLAB_BYTES / 16; i += 256)
    ((i32x4*)dst)[i] = ((const i32x4*)tile)[i];
}

// ---------------------------------------------------------------------------
// Shared WMMA mainloop + epilogue (LDS tile already holds 3 padded rows)
// ---------------------------------------------------------------------------
__device__ __forceinline__ void conv_mainloop(
    const unsigned short* __restrict__ ldsX,
    const unsigned short* __restrict__ Wb,
    const float* __restrict__ bias,
    float* __restrict__ out, int b, int y, int mhalf) {
  const int lane = threadIdx.x & 31;
  const int wave = threadIdx.x >> 5;
  const int mq   = wave >> 2;       // 0..1  -> 64 oc rows each
  const int nq   = wave & 3;        // 0..3  -> 16 output cols each
  const int lhi  = lane >> 4;       // which K-half this lane carries
  const int llo  = lane & 15;

  v8f acc[4];
  const v8f vzero = {0.f, 0.f, 0.f, 0.f, 0.f, 0.f, 0.f, 0.f};
#pragma unroll
  for (int f = 0; f < 4; ++f) acc[f] = vzero;

  const int ocWave = mhalf * 128 + mq * 64;

  for (int tap = 0; tap < KTAPS; ++tap) {
    const int ky = tap / 3;
    const int kx = tap - ky * 3;
    const unsigned short* ldsRow = ldsX + ky * SLAB_ELEMS;
    const unsigned short* Wt     = Wb + (size_t)tap * (256 * 256);

#pragma unroll
    for (int kc = 0; kc < 256; kc += 32) {
      // B fragment: 32(ch) x 16(cols); lane = col, 16 consecutive channels.
      Frag bf;
      const unsigned short* bp =
          ldsRow + (nq * 16 + llo + kx) * XSTRIDE + kc + lhi * 16;
      bf.q[0] = *(const i32x4*)(bp);
      bf.q[1] = *(const i32x4*)(bp + 8);

      __builtin_prefetch(Wt + (ocWave + llo) * 256 + kc + 32, 0, 1);

#pragma unroll
      for (int f = 0; f < 4; ++f) {
        const int ocRow = ocWave + f * 16 + llo;
        const unsigned short* ap = Wt + ocRow * 256 + kc + lhi * 16;
        Frag af;
        af.q[0] = *(const i32x4*)(ap);        // K lhi*16 .. +7
        af.q[1] = *(const i32x4*)(ap + 8);    // K lhi*16+8 .. +15
        acc[f] = __builtin_amdgcn_wmma_f32_16x16x32_bf16(
            false, af.v, false, bf.v, (short)0, acc[f], false, false);
      }
    }
  }

  float* outp = out + (size_t)b * IMG + (size_t)y * HW;
  const int xcol = nq * 16 + llo;
#pragma unroll
  for (int f = 0; f < 4; ++f) {
    const int ocBase = ocWave + f * 16 + lhi * 8;
#pragma unroll
    for (int v = 0; v < 8; ++v) {
      const int oc = ocBase + v;
      outp[(size_t)oc * (HW * HW) + xcol] = acc[f][v] + bias[oc >> 2];
    }
  }
}

// ---------------------------------------------------------------------------
// Stage 3 (fast): stage 104544 contiguous bytes of Xpad -> LDS with the
// Tensor Data Mover (1-D descriptor, 13068 x 8B), then WMMA.
// ---------------------------------------------------------------------------
__global__ __launch_bounds__(256) void conv_wmma_tdm_kernel(
    const unsigned short* __restrict__ Xpad,   // [32][66][66][264] bf16
    const unsigned short* __restrict__ Wb,     // [9][256][256] bf16
    const float* __restrict__ bias,
    float* __restrict__ out) {
  extern __shared__ unsigned short ldsX[];     // [3][66][XSTRIDE] bf16

  const int bid   = blockIdx.x;
  const int mhalf = bid & 1;
  const int rowid = bid >> 1;
  const int b     = rowid >> 6;
  const int y     = rowid & 63;

  // 3 padded rows y..y+2 are one contiguous range of Xpad
  const unsigned short* slab = Xpad + ((size_t)b * PADH + y) * SLAB_ELEMS;

#if __has_builtin(__builtin_amdgcn_tensor_load_to_lds)
  if (threadIdx.x < 32) {          // one wave issues the TDM transfer
    const unsigned long long ga = (unsigned long long)(const void*)slab;
    const unsigned n8 = LDS_BYTES / 8;          // 13068 8-byte elements
    u32x4 g0;
    g0[0] = 1u;                                  // count=1, user mode
    g0[1] = 0u;                                  // lds_addr = 0 (dynamic LDS base)
    g0[2] = (unsigned)(ga & 0xffffffffu);        // global_addr[31:0]
    g0[3] = (unsigned)((ga >> 32) & 0x1ffffffu) | (2u << 30);  // [56:32] | type=2
    i32x8 g1;
    g1[0] = 0x00030000;                          // wg_mask=0, data_size=8B
    g1[1] = (int)((n8 & 0xffffu) << 16);         // tensor_dim0[15:0]
    g1[2] = (int)(((n8 >> 16) & 0xffffu) | (1u << 16));  // dim0 hi | tensor_dim1=1
    g1[3] = (int)((n8 & 0xffffu) << 16);         // tile_dim0 = n8
    g1[4] = 1;                                   // tile_dim1=1, tile_dim2=0
    g1[5] = (int)n8;                             // tensor_dim0_stride lo
    g1[6] = 0;
    g1[7] = 0;
    const i32x4 gz = {0, 0, 0, 0};
#if defined(__clang_major__) && (__clang_major__ >= 23)
    const i32x8 gz8 = {0, 0, 0, 0, 0, 0, 0, 0};
    __builtin_amdgcn_tensor_load_to_lds(g0, g1, gz, gz, gz8, 0);
#else
    __builtin_amdgcn_tensor_load_to_lds(g0, g1, gz, gz, 0);
#endif
#if __has_builtin(__builtin_amdgcn_s_wait_tensorcnt)
    __builtin_amdgcn_s_wait_tensorcnt(0);
#else
    asm volatile("s_wait_tensorcnt 0x0" ::: "memory");
#endif
  }
#else
  // fallback: plain vectorized copy (still a pure b128 memcpy, no convert)
  for (int i = threadIdx.x; i < LDS_BYTES / 16; i += 256)
    ((i32x4*)ldsX)[i] = ((const i32x4*)slab)[i];
#endif
  __syncthreads();

  conv_mainloop(ldsX, Wb, bias, out, b, y, mhalf);
}

// ---------------------------------------------------------------------------
// Stage 3 (fallback when workspace too small): round-1 style conv that
// converts fp32 x -> bf16 LDS tile in-kernel.
// ---------------------------------------------------------------------------
__global__ __launch_bounds__(256) void conv_wmma_fallback_kernel(
    const float* __restrict__ x,
    const unsigned short* __restrict__ Wb,
    const float* __restrict__ bias,
    float* __restrict__ out) {
  extern __shared__ unsigned short ldsX[];

  const int bid   = blockIdx.x;
  const int mhalf = bid & 1;
  const int rowid = bid >> 1;
  const int b     = rowid >> 6;
  const int y     = rowid & 63;
  const float* xb = x + (size_t)b * IMG;

  for (int task = threadIdx.x; task < 3 * 256; task += 256) {
    const int r  = task >> 8;
    const int c  = task & 255;
    const int yy = y - 1 + r;
    unsigned short* drow = ldsX + r * SLAB_ELEMS + c;
    if (yy < 0 || yy >= HW) {
      for (int p = 0; p < PADH; ++p) drow[p * XSTRIDE] = 0;
    } else {
      const float* src = xb + c * (HW * HW) + yy * HW;
      drow[0] = 0;
      drow[(PADH - 1) * XSTRIDE] = 0;
#pragma unroll
      for (int xx = 0; xx < HW; xx += 4) {
        f32x4 v4 = *(const f32x4*)(src + xx);
        drow[(xx + 1) * XSTRIDE] = f2bf(v4.x);
        drow[(xx + 2) * XSTRIDE] = f2bf(v4.y);
        drow[(xx + 3) * XSTRIDE] = f2bf(v4.z);
        drow[(xx + 4) * XSTRIDE] = f2bf(v4.w);
      }
    }
  }
  __syncthreads();

  conv_mainloop(ldsX, Wb, bias, out, b, y, mhalf);
}

// ---------------------------------------------------------------------------
// Host launcher
// ---------------------------------------------------------------------------
extern "C" void kernel_launch(void* const* d_in, const int* in_sizes, int n_in,
                              void* d_out, int out_size, void* d_ws, size_t ws_size,
                              hipStream_t stream) {
  const float* x      = (const float*)d_in[0];  // [32][64][4][64][64]
  const float* weight = (const float*)d_in[1];  // [64][64][36]
  const float* reps   = (const float*)d_in[2];  // [4][36][36]
  const float* bias   = (const float*)d_in[3];  // [64]
  float* out = (float*)d_out;

  float*          repSink = (float*)d_ws;
  unsigned short* Wb   = (unsigned short*)((char*)d_ws + WB_OFF);
  unsigned short* Xpad = (unsigned short*)((char*)d_ws + XPAD_OFF);

  (void)in_sizes; (void)n_in; (void)out_size;

  hipFuncSetAttribute((const void*)conv_wmma_tdm_kernel,
                      hipFuncAttributeMaxDynamicSharedMemorySize, LDS_BYTES);
  hipFuncSetAttribute((const void*)conv_wmma_fallback_kernel,
                      hipFuncAttributeMaxDynamicSharedMemorySize, LDS_BYTES);

  sinkhorn_kernel<<<dim3(4), dim3(64), 0, stream>>>(reps, repSink);
  build_weights_kernel<<<dim3(256), dim3(256), 0, stream>>>(repSink, weight, Wb);

  const bool bigWs = ws_size >= (size_t)XPAD_OFF + XPAD_BYTES;
  if (bigWs) {
    pad_x_bf16_kernel<<<dim3(BATCH * PADH), dim3(256), 0, stream>>>(x, Xpad);
    conv_wmma_tdm_kernel<<<dim3(BATCH * HW * 2), dim3(256), LDS_BYTES, stream>>>(
        Xpad, Wb, bias, out);
  } else {
    conv_wmma_fallback_kernel<<<dim3(BATCH * HW * 2), dim3(256), LDS_BYTES, stream>>>(
        x, Wb, bias, out);
  }
}
